// Attention_9577777070769
// MI455X (gfx1250) — matrix operand order repaired and checked by reading, not compiled, not run on hardware
//
#include <hip/hip_runtime.h>
#include <hip/hip_bf16.h>

typedef __attribute__((ext_vector_type(16))) __bf16 v16bf;
typedef __attribute__((ext_vector_type(8)))  __bf16 v8bf;
typedef __attribute__((ext_vector_type(8)))  float  v8f;

#define BB 32
#define TKK 1024
#define NN 1024
#define ADK 512

// ---------------------------------------------------------------------------
// Kernel P: pack W_r_w (fp32 [N, AD]) -> bf16 once; shared by all 2048 tiles.
// The GEMM then loads B fragments as raw 32B bf16 vectors: no cvt in hot loop.
// ---------------------------------------------------------------------------
__global__ __launch_bounds__(256) void pack_w_kernel(
    const float* __restrict__ W, __bf16* __restrict__ Wbf)
{
    const int i = (blockIdx.x * 256 + threadIdx.x) * 4;   // 4 elems/thread
    const float4 w = *(const float4*)(W + i);
    Wbf[i + 0] = (__bf16)w.x;
    Wbf[i + 1] = (__bf16)w.y;
    Wbf[i + 2] = (__bf16)w.z;
    Wbf[i + 3] = (__bf16)w.w;
}

// ---------------------------------------------------------------------------
// Kernel 0: zero struct-score accumulator + compute dec_fea = s_t_hat@W_s^T + b
// ---------------------------------------------------------------------------
__global__ __launch_bounds__(256) void init_decfea_kernel(
    const float* __restrict__ s_t_hat, const float* __restrict__ W_s_w,
    const float* __restrict__ W_s_b, float* __restrict__ dec_fea,
    float* __restrict__ s2_accum)
{
    const int id = blockIdx.x * 256 + threadIdx.x;   // 0..32767
    s2_accum[id] = 0.0f;
    const int b = id >> 10;
    const int n = id & (NN - 1);
    const float4* sr = (const float4*)(s_t_hat + (size_t)b * NN);
    const float4* wr = (const float4*)(W_s_w + (size_t)n * NN);
    float acc = 0.0f;
    for (int k = 0; k < NN / 4; ++k) {
        float4 a = sr[k], w = wr[k];
        acc += a.x * w.x + a.y * w.y + a.z * w.z + a.w * w.w;
    }
    dec_fea[id] = acc + W_s_b[n];
}

// ---------------------------------------------------------------------------
// Kernel 1: attention-branch scores (streams encoder_feature, 128 MB)
// One wave32 per (b,t) row.
// ---------------------------------------------------------------------------
__global__ __launch_bounds__(256) void attn_scores_kernel(
    const float* __restrict__ enc_feat, const float* __restrict__ dec_fea,
    const float* __restrict__ coverage, const float* __restrict__ W_c_w,
    const float* __restrict__ v_w, float* __restrict__ scores)
{
    const int wid  = (blockIdx.x * 256 + threadIdx.x) >> 5;
    const int lane = threadIdx.x & 31;
    const int b = wid >> 10;
    const int t = wid & (TKK - 1);
    const float cov = coverage[b * TKK + t];
    const float4* er = (const float4*)(enc_feat + ((size_t)(b * TKK + t)) * NN);
    const float4* dr = (const float4*)(dec_fea + (size_t)b * NN);
    const float4* wc = (const float4*)W_c_w;
    const float4* vv = (const float4*)v_w;
    float sum = 0.0f;
    #pragma unroll
    for (int i = 0; i < 8; ++i) {
        const int idx = i * 32 + lane;
        float4 e = er[idx], d = dr[idx], c = wc[idx], v = vv[idx];
        sum += tanhf(e.x + d.x + cov * c.x) * v.x;
        sum += tanhf(e.y + d.y + cov * c.y) * v.y;
        sum += tanhf(e.z + d.z + cov * c.z) * v.z;
        sum += tanhf(e.w + d.w + cov * c.w) * v.w;
    }
    #pragma unroll
    for (int off = 16; off; off >>= 1) sum += __shfl_xor(sum, off);
    if (lane == 0) scores[b * TKK + t] = sum;
}

// ---------------------------------------------------------------------------
// Kernel 2: structural-branch scores via bf16 WMMA, fused epilogue.
// A tile (r_i[b, t0..t0+15, :]) is staged into LDS with CDNA5 async
// global->LDS copies (ASYNCcnt), converted to bf16 in LDS, then the hot loop
// is: 2x ds_load_b128 (A frag) + per n-tile 2x global b128 (bf16 B frag)
// + v_wmma_f32_16x16x32_bf16.
// ---------------------------------------------------------------------------
__global__ __launch_bounds__(256) void struct_scores_kernel(
    const float* __restrict__ r_i, const __bf16* __restrict__ Wbf,
    const float* __restrict__ dec_fea, const float* __restrict__ v_w,
    float* __restrict__ s2_accum)
{
    __shared__ __align__(16) float  Araw[16][ADK];       // 32 KB fp32 staging
    __shared__ __align__(16) __bf16 Atile[16][ADK + 8];  // padded: conflict-free

    const int b  = blockIdx.x >> 6;
    const int t0 = (blockIdx.x & 63) * 16;
    const int tid = threadIdx.x;

    // --- async copy r_i tile (16x512 fp32 = 32KB) into LDS: 8 b128 per lane ---
    {
        const unsigned int lds_base = (unsigned int)(uintptr_t)(&Araw[0][0]);
        #pragma unroll
        for (int i = 0; i < 8; ++i) {
            const int c   = tid + 256 * i;        // 16-byte chunk index
            const int row = c >> 7;
            const int col = (c & 127) * 4;
            const unsigned long long gaddr = (unsigned long long)(uintptr_t)
                (r_i + ((size_t)(b * TKK + t0 + row)) * ADK + col);
            const unsigned int laddr = lds_base + (unsigned int)c * 16u;
            asm volatile("global_load_async_to_lds_b128 %0, %1, off"
                         :: "v"(laddr), "v"(gaddr) : "memory");
        }
        asm volatile("s_wait_asynccnt 0x0" ::: "memory");
    }
    __syncthreads();

    // --- LDS fp32 -> LDS bf16 (padded rows for ds_load_b128) ---
    for (int i = tid; i < 16 * ADK; i += 256) {
        const int row = i >> 9, a = i & (ADK - 1);
        Atile[row][a] = (__bf16)Araw[row][a];
    }
    __syncthreads();

    const int wave  = tid >> 5;
    const int lane  = tid & 31;
    const int m16   = lane & 15;       // row (A) / column (B,D) within tile
    const int khalf = lane >> 4;       // which K-half this lane holds

    v8f acc[8];
    #pragma unroll
    for (int q = 0; q < 8; ++q) acc[q] = (v8f){};

    #pragma unroll 1
    for (int ks = 0; ks < ADK / 32; ++ks) {
        const int kb = ks * 32;
        // A fragment (16x32 bf16): two 16B LDS vector loads
        const v8bf alo = *(const v8bf*)&Atile[m16][kb + khalf * 16];
        const v8bf ahi = *(const v8bf*)&Atile[m16][kb + khalf * 16 + 8];
        const v16bf afrag = __builtin_shufflevector(
            alo, ahi, 0, 1, 2, 3, 4, 5, 6, 7, 8, 9, 10, 11, 12, 13, 14, 15);

        #pragma unroll
        for (int q = 0; q < 8; ++q) {
            const int ncol = (wave + q * 8) * 16 + m16;
            // B fragment: 16 consecutive bf16 K-values, one 32B vector load
            const v16bf bfrag =
                *(const v16bf*)(Wbf + (size_t)ncol * ADK + kb + khalf * 16);
            acc[q] = __builtin_amdgcn_wmma_f32_16x16x32_bf16(
                false, afrag, false, bfrag, (short)0, acc[q], false, false);
        }
    }

    // Fused epilogue: tanh(D + dec_fea) * v, reduce over the 16 N-columns,
    // accumulate per t-row into global scores.
    float rsum[8];
    #pragma unroll
    for (int j = 0; j < 8; ++j) rsum[j] = 0.0f;

    #pragma unroll
    for (int q = 0; q < 8; ++q) {
        const int ncol = (wave + q * 8) * 16 + m16;
        const float dv = dec_fea[b * NN + ncol];
        const float vv = v_w[ncol];
        #pragma unroll
        for (int j = 0; j < 8; ++j) {
            // D layout: VGPR j = row j (lanes 0-15) and row j+8 (lanes 16-31)
            float e = tanhf(acc[q][j] + dv) * vv;
            #pragma unroll
            for (int off = 8; off; off >>= 1) e += __shfl_xor(e, off, 16);
            rsum[j] += e;
        }
    }
    if (m16 == 0) {
        #pragma unroll
        for (int j = 0; j < 8; ++j)
            atomicAdd(&s2_accum[b * TKK + t0 + j + khalf * 8], rsum[j]);
    }
}

// ---------------------------------------------------------------------------
// Kernel 3: both softmaxes. One block (256 thr) per batch row.
// attn renormalization cancels: attn = e^{s-M}*mask / sum(e^{s-M}*mask)
// ---------------------------------------------------------------------------
__global__ __launch_bounds__(256) void softmax_kernel(
    const float* __restrict__ scores, const float* __restrict__ s2,
    const float* __restrict__ mask, const float* __restrict__ coverage,
    float* __restrict__ attn, float* __restrict__ cov_new,
    float* __restrict__ astruct)
{
    const int b = blockIdx.x, tid = threadIdx.x;
    const int lane = tid & 31, wid = tid >> 5;
    __shared__ float sred[8];

    const float4 mk = ((const float4*)(mask + b * TKK))[tid];

    // ---- branch 1 ----
    float4 s = ((const float4*)(scores + b * TKK))[tid];
    float m = fmaxf(fmaxf(s.x, s.y), fmaxf(s.z, s.w));
    #pragma unroll
    for (int off = 16; off; off >>= 1) m = fmaxf(m, __shfl_xor(m, off));
    if (lane == 0) sred[wid] = m;
    __syncthreads();
    float M = sred[0];
    #pragma unroll
    for (int i = 1; i < 8; ++i) M = fmaxf(M, sred[i]);
    __syncthreads();

    float4 p;
    p.x = __expf(s.x - M) * mk.x; p.y = __expf(s.y - M) * mk.y;
    p.z = __expf(s.z - M) * mk.z; p.w = __expf(s.w - M) * mk.w;
    float ls = p.x + p.y + p.z + p.w;
    #pragma unroll
    for (int off = 16; off; off >>= 1) ls += __shfl_xor(ls, off);
    if (lane == 0) sred[wid] = ls;
    __syncthreads();
    float S = 0.0f;
    #pragma unroll
    for (int i = 0; i < 8; ++i) S += sred[i];
    __syncthreads();
    const float inv = 1.0f / S;
    float4 a; a.x = p.x * inv; a.y = p.y * inv; a.z = p.z * inv; a.w = p.w * inv;
    ((float4*)(attn + b * TKK))[tid] = a;
    float4 cv = ((const float4*)(coverage + b * TKK))[tid];
    cv.x += a.x; cv.y += a.y; cv.z += a.z; cv.w += a.w;
    ((float4*)(cov_new + b * TKK))[tid] = cv;

    // ---- branch 2 ----
    float4 s2v = ((const float4*)(s2 + b * TKK))[tid];
    float m2 = fmaxf(fmaxf(s2v.x, s2v.y), fmaxf(s2v.z, s2v.w));
    #pragma unroll
    for (int off = 16; off; off >>= 1) m2 = fmaxf(m2, __shfl_xor(m2, off));
    if (lane == 0) sred[wid] = m2;
    __syncthreads();
    float M2 = sred[0];
    #pragma unroll
    for (int i = 1; i < 8; ++i) M2 = fmaxf(M2, sred[i]);
    __syncthreads();

    float4 p2;
    p2.x = __expf(s2v.x - M2); p2.y = __expf(s2v.y - M2);
    p2.z = __expf(s2v.z - M2); p2.w = __expf(s2v.w - M2);
    float ls2 = p2.x + p2.y + p2.z + p2.w;
    #pragma unroll
    for (int off = 16; off; off >>= 1) ls2 += __shfl_xor(ls2, off);
    if (lane == 0) sred[wid] = ls2;
    __syncthreads();
    float S2 = 0.0f;
    #pragma unroll
    for (int i = 0; i < 8; ++i) S2 += sred[i];
    const float inv2 = 1.0f / S2;
    float4 as;
    as.x = p2.x * inv2 * mk.x; as.y = p2.y * inv2 * mk.y;
    as.z = p2.z * inv2 * mk.z; as.w = p2.w * inv2 * mk.w;
    ((float4*)(astruct + b * TKK))[tid] = as;
}

// ---------------------------------------------------------------------------
// Kernel 4: both contexts in ONE pass over encoder_outputs (128 MB).
// ---------------------------------------------------------------------------
__global__ __launch_bounds__(256) void context_kernel(
    const float* __restrict__ eo, const float* __restrict__ attn,
    const float* __restrict__ astruct, float* __restrict__ c_t,
    float* __restrict__ c_ts)
{
    __shared__ float wa[TKK];
    __shared__ float ws[TKK];
    const int b = blockIdx.y;
    const int n = blockIdx.x * 256 + threadIdx.x;
    for (int i = threadIdx.x; i < TKK; i += 256) {
        wa[i] = attn[b * TKK + i];
        ws[i] = astruct[b * TKK + i];
    }
    __syncthreads();
    const float* base = eo + (size_t)b * TKK * NN + n;
    float a0 = 0.0f, a1 = 0.0f;
    #pragma unroll 4
    for (int t = 0; t < TKK; ++t) {
        const float x = base[(size_t)t * NN];
        a0 += wa[t] * x;
        a1 += ws[t] * x;
    }
    c_t[b * NN + n]  = a0;
    c_ts[b * NN + n] = a1;
}

// ---------------------------------------------------------------------------
extern "C" void kernel_launch(void* const* d_in, const int* in_sizes, int n_in,
                              void* d_out, int out_size, void* d_ws, size_t ws_size,
                              hipStream_t stream)
{
    const float* s_t_hat    = (const float*)d_in[0];
    const float* enc_out    = (const float*)d_in[1];
    const float* enc_feat   = (const float*)d_in[2];
    const float* pad_mask   = (const float*)d_in[3];
    const float* coverage   = (const float*)d_in[4];
    const float* r_i        = (const float*)d_in[5];
    const float* W_s_w      = (const float*)d_in[6];
    const float* W_s_b      = (const float*)d_in[7];
    const float* W_c_w      = (const float*)d_in[8];
    const float* v_w        = (const float*)d_in[9];
    const float* W_r_w      = (const float*)d_in[10];

    float* out      = (float*)d_out;
    float* c_t      = out;                 // [32,1024]
    float* attn     = out + 32768;         // [32,1024]
    float* cov_new  = out + 65536;         // [32,1024]
    float* c_ts     = out + 98304;         // [32,1024]
    float* astruct  = out + 131072;        // [32,1,1024]

    float*  ws       = (float*)d_ws;
    float*  dec_fea  = ws;                 // [32,1024]
    float*  scores   = ws + 32768;         // [32,1024]
    float*  s2_accum = ws + 65536;         // [32,1024]
    __bf16* Wbf      = (__bf16*)(ws + 98304);  // [1024,512] bf16 (1 MB)

    pack_w_kernel<<<512, 256, 0, stream>>>(W_r_w, Wbf);
    init_decfea_kernel<<<128, 256, 0, stream>>>(s_t_hat, W_s_w, W_s_b,
                                                dec_fea, s2_accum);
    attn_scores_kernel<<<4096, 256, 0, stream>>>(enc_feat, dec_fea, coverage,
                                                 W_c_w, v_w, scores);
    struct_scores_kernel<<<BB * 64, 256, 0, stream>>>(r_i, Wbf, dec_fea,
                                                      v_w, s2_accum);
    softmax_kernel<<<BB, 256, 0, stream>>>(scores, s2_accum, pad_mask, coverage,
                                           attn, cov_new, astruct);
    context_kernel<<<dim3(NN / 256, BB), 256, 0, stream>>>(enc_out, attn,
                                                           astruct, c_t, c_ts);
}
